// PlainDecoder_28071906246896
// MI455X (gfx1250) — matrix operand order repaired and checked by reading, not compiled, not run on hardware
//
#include <hip/hip_runtime.h>
#include <math.h>

// ---------------- problem constants ----------------
#define B_     128
#define N_     64
#define M_     196
#define D_     512
#define H_     8
#define HD_    64
#define FF_    2048
#define VOC_   10000
#define DEPTH_ 3
#define TOK_   (B_ * N_)    // 8192 decoder tokens
#define ETOK_  (B_ * M_)    // 25088 encoder tokens
#define SCALE_ 0.125f       // HD^-0.5
#define MPMAX_ 224          // max kv length padded to 32

typedef __attribute__((ext_vector_type(16))) __bf16 v16bf;
typedef __attribute__((ext_vector_type(2)))  __bf16 v2bf;
typedef __attribute__((ext_vector_type(8)))  float  v8f;
typedef int          v4i  __attribute__((vector_size(16)));  // async builtin param
typedef unsigned int v4u  __attribute__((vector_size(16)));  // TDM group0
typedef int          v8i_ __attribute__((vector_size(32)));  // TDM group1
typedef int          v4i_ __attribute__((vector_size(16)));  // TDM group2/3

union FragU { v16bf v; unsigned u[8]; };

// feature gates (device pass only; host pass sees fallbacks)
#if defined(__gfx1250__) && __has_builtin(__builtin_amdgcn_global_load_async_to_lds_b128)
#define ASYNC_OK 1
#else
#define ASYNC_OK 0
#endif

#if defined(__gfx1250__) && __has_builtin(__builtin_amdgcn_cvt_pk_bf16_f32)
#define CVT_PK_OK 1
#else
#define CVT_PK_OK 0
#endif

#if defined(__gfx1250__) && __has_builtin(__builtin_amdgcn_tensor_load_to_lds)
#define TDM_OK 1
#else
#define TDM_OK 0
#endif

#define GLOBAL_AS __attribute__((address_space(1)))
#define LDS_AS    __attribute__((address_space(3)))

__device__ __forceinline__ GLOBAL_AS v4i* as_global_v4(const void* p) {
  return (GLOBAL_AS v4i*)(unsigned long long)(size_t)p;
}
__device__ __forceinline__ LDS_AS v4i* as_lds_v4(void* p) {
  return (LDS_AS v4i*)(unsigned)(size_t)p;
}

// pack two f32 -> packed bf16 (low = a, high = b): single VALU instruction
__device__ __forceinline__ unsigned pack2bf(float a, float b) {
#if CVT_PK_OK
  union { v2bf v; unsigned u; } c;
  c.v = __builtin_amdgcn_cvt_pk_bf16_f32(a, b);   // v_cvt_pk_bf16_f32 (RNE)
  return c.u;
#else
  // one v_perm_b32: take high 16 bits of each (truncation)
  return __builtin_amdgcn_perm(__float_as_uint(b), __float_as_uint(a), 0x07060302u);
#endif
}
__device__ __forceinline__ unsigned short f2bf(float f) {
  return (unsigned short)(pack2bf(f, 0.0f) & 0xFFFFu);
}

// A fragment (16x32 bf16, M x K) from bf16 LDS tile: lane holds row (lane&15)
// VGPR v: K = (lane>=16)*16 + 2v + {0,1}
__device__ __forceinline__ v16bf load_a_frag(const unsigned short* base, int row,
                                             int stride, int kk, int lane) {
  const unsigned short* p = base + row * stride + kk;
  const int half16 = (lane >> 4) << 4;
  FragU f;
#pragma unroll
  for (int v = 0; v < 8; ++v) {
    int k = half16 + (v << 1);
    f.u[v] = *(const unsigned*)(p + k);
  }
  return f.v;
}

// B fragment (32x16 bf16, K x N) from transposed bf16 tile Bt[n][k]:
// lane holds column (lane&15); VGPR v: K = (lane>=16)*16 + 2v + {0,1}
__device__ __forceinline__ v16bf load_b_frag(const unsigned short* base, int col,
                                             int stride, int kk, int lane) {
  const unsigned short* p = base + col * stride + kk + ((lane >> 4) << 4);
  FragU f;
#pragma unroll
  for (int v = 0; v < 8; ++v) f.u[v] = *(const unsigned*)(p + (v << 1));
  return f.v;
}

#if TDM_OK
// Tensor Data Mover: DMA a fully in-bounds 2D fp32 tile into contiguous LDS.
// D# packing per CDNA5 ISA ch.8 (groups 0/1; groups 2/3 zero for 2-D tensors).
// tile rows land back-to-back in LDS (tile_d0*4 bytes per row).
__device__ __forceinline__ void tdm_load_2d_f32(
    const void* gptr, void* lptr,
    unsigned tile_d0 /*elems per row*/, unsigned tile_d1 /*rows*/,
    unsigned stride_d0 /*row stride, elems*/) {
  unsigned long long ga = (unsigned long long)(size_t)gptr;
  unsigned lds = (unsigned)(size_t)lptr;
  v4u g0;
  g0[0] = 1u;                                         // count=1 (user descriptor)
  g0[1] = lds;                                        // lds_addr (bytes)
  g0[2] = (unsigned)(ga & 0xFFFFFFFFu);               // global_addr[31:0]
  g0[3] = (unsigned)((ga >> 32) & 0x01FFFFFFu)        // global_addr[56:32]
        | (2u << 30);                                 // type=2 ("image")
  v8i_ g1;
  g1[0] = (int)(2u << 16);                            // data_size=2 (4 bytes)
  g1[1] = (int)((tile_d0 & 0xFFFFu) << 16);           // tensor_dim0[15:0] @63:48
  g1[2] = (int)((tile_d0 >> 16) |                     // tensor_dim0[31:16]
                ((tile_d1 & 0xFFFFu) << 16));         // tensor_dim1[15:0]
  g1[3] = (int)((tile_d1 >> 16) |                     // tensor_dim1[31:16]
                (tile_d0 << 16));                     // tile_dim0
  g1[4] = (int)(tile_d1 & 0xFFFFu);                   // tile_dim1; tile_dim2=0
  g1[5] = (int)stride_d0;                             // tensor_dim0_stride[31:0]
  g1[6] = 0;                                          // stride hi / dim1 stride
  g1[7] = 0;
  v4i_ gz = {0, 0, 0, 0};
#if defined(__clang_major__) && (__clang_major__ >= 23)
  v8i_ gz8 = {0, 0, 0, 0, 0, 0, 0, 0};
  __builtin_amdgcn_tensor_load_to_lds(g0, g1, gz, gz, gz8, 0);
#else
  __builtin_amdgcn_tensor_load_to_lds(g0, g1, gz, gz, 0);
#endif
}
#endif

// =======================================================================
// GEMM: C[M,N] = act( A[M,K] @ W[K,N] + bias[N] )
// bf16 WMMA compute; fp32 tiles staged in LDS.
// Hot path: A tile via per-lane async DMA (ASYNCcnt), W tile via TDM
// descriptor (TENSORcnt). block = 256 threads, tile 128x128x32, 2 buffers.
// =======================================================================

// one K-step of compute from fp32 LDS tiles (convert-to-bf16 in registers)
__device__ __forceinline__ void gemm_step(const float* __restrict__ Asf,
                                          const float* __restrict__ Wsf,
                                          v8f acc[4][2], int wm, int wn, int lane) {
  v16bf bfrag[2];
#pragma unroll
  for (int tj = 0; tj < 2; ++tj) {
    int n = wn * 32 + tj * 16 + (lane & 15);
    int kbase = (lane >> 4) << 4;          // 0 or 16
    FragU f;
#pragma unroll
    for (int v = 0; v < 8; ++v) {
      int k = kbase + 2 * v;
      f.u[v] = pack2bf(Wsf[k * 128 + n], Wsf[(k + 1) * 128 + n]);
    }
    bfrag[tj] = f.v;
  }
  const int half8 = (lane >> 4) << 3;
#pragma unroll
  for (int ti = 0; ti < 4; ++ti) {
    int row = wm * 64 + ti * 16 + (lane & 15);
    FragU fa;
#pragma unroll
    for (int v = 0; v < 8; ++v) {
      int k = (half8 << 1) + (v << 1);
      float2 t = *(const float2*)&Asf[row * 32 + k];
      fa.u[v] = pack2bf(t.x, t.y);
    }
#pragma unroll
    for (int tj = 0; tj < 2; ++tj) {
      acc[ti][tj] = __builtin_amdgcn_wmma_f32_16x16x32_bf16(
          false, fa.v, false, bfrag[tj], (short)0, acc[ti][tj], false, false);
    }
  }
}

#if ASYNC_OK
// stage one K-slice: A 128x32 f32 via per-lane async b128 (4 instrs/wave),
// W 32x128 f32 via one TDM descriptor from wave 0 (or async fallback).
__device__ __forceinline__ void issue_tiles(
    const float* __restrict__ A, const float* __restrict__ W,
    int K, int Ncols, int mBase, int nBase, int kk,
    float* AsBuf, float* WsBuf, int tid) {
  const int lane = tid & 31;
  const int w    = tid >> 5;
  // A tile: lane covers 16B; 4 rows per instruction (lane>>3)
#pragma unroll
  for (int j = 0; j < 4; ++j) {
    int r  = (w * 4 + j) * 4 + (lane >> 3);      // 0..127
    int cb = (lane & 7) * 16;                    // byte offset within row
    const float* gp = A + (size_t)(mBase + r) * K + kk;
    __builtin_amdgcn_global_load_async_to_lds_b128(
        as_global_v4((const char*)gp + cb),
        as_lds_v4((char*)&AsBuf[r * 32] + cb), 0, 0);
  }
#if TDM_OK
  // W tile: one TDM 2D descriptor, issued by wave 0 only
  if (w == 0) {
    tdm_load_2d_f32(W + (size_t)kk * (size_t)Ncols + nBase, WsBuf,
                    128u, 32u, (unsigned)Ncols);
  }
#else
  // W tile: one 512B row per async instruction, 4 instrs/wave
#pragma unroll
  for (int j = 0; j < 4; ++j) {
    int k = w * 4 + j;                           // 0..31
    const float* gp = W + (size_t)(kk + k) * (size_t)Ncols + nBase;
    __builtin_amdgcn_global_load_async_to_lds_b128(
        as_global_v4((const char*)gp + lane * 16),
        as_lds_v4((char*)&WsBuf[k * 128] + lane * 16), 0, 0);
  }
#endif
}
#endif

// guarded synchronous staging (tail N-blocks and non-async fallback)
__device__ __forceinline__ void stage_tiles_sync(
    const float* __restrict__ A, const float* __restrict__ W,
    int K, int Ncols, int Mrows, int mBase, int nBase, int kk,
    float* AsBuf, float* WsBuf, int tid) {
#pragma unroll
  for (int it = 0; it < 4; ++it) {               // A: 1024 float4
    int idx = tid + it * 256;
    int r = idx >> 3, c4 = (idx & 7) * 4;
    int grow = mBase + r;
    float4 f;
    if (grow < Mrows) f = *(const float4*)(A + (size_t)grow * K + kk + c4);
    else { f.x = f.y = f.z = f.w = 0.f; }
    *(float4*)&AsBuf[r * 32 + c4] = f;
  }
#pragma unroll
  for (int it = 0; it < 4; ++it) {               // W: 1024 float4
    int idx = tid + it * 256;
    int k = idx >> 5, c4 = (idx & 31) * 4;
    int gc = nBase + c4;
    const float* wrow = W + (size_t)(kk + k) * (size_t)Ncols;
    float4 f;
    if (gc + 3 < Ncols) {
      f = *(const float4*)(wrow + gc);
    } else {
      f.x = (gc + 0 < Ncols) ? wrow[gc + 0] : 0.f;
      f.y = (gc + 1 < Ncols) ? wrow[gc + 1] : 0.f;
      f.z = (gc + 2 < Ncols) ? wrow[gc + 2] : 0.f;
      f.w = (gc + 3 < Ncols) ? wrow[gc + 3] : 0.f;
    }
    *(float4*)&WsBuf[k * 128 + c4] = f;
  }
}

__global__ __launch_bounds__(256) void gemm_bf16_kernel(
    const float* __restrict__ A, const float* __restrict__ W,
    const float* __restrict__ bias, float* __restrict__ C,
    int Mrows, int K, int Ncols, int relu) {
  __shared__ float Asf[2][128 * 32];   // 32 KB
  __shared__ float Wsf[2][32 * 128];   // 32 KB

  const int tid  = threadIdx.x;
  const int lane = tid & 31;
  const int w    = tid >> 5;
  const int wm   = w >> 2;      // 0..1  (64 rows each)
  const int wn   = w & 3;       // 0..3  (32 cols each)
  const int mBase = blockIdx.y * 128;
  const int nBase = blockIdx.x * 128;

  v8f zero = {};
  v8f acc[4][2];
#pragma unroll
  for (int i = 0; i < 4; ++i)
#pragma unroll
    for (int j = 0; j < 2; ++j) acc[i][j] = zero;

  const int nsteps = K >> 5;    // K is a multiple of 32 for all call sites

#if ASYNC_OK
  const bool tailN = (nBase + 128 > Ncols);
  if (!tailN) {
    // ---- async + TDM double-buffered pipeline ----
    issue_tiles(A, W, K, Ncols, mBase, nBase, 0, Asf[0], Wsf[0], tid);
    for (int s = 0; s < nsteps; ++s) {
      int cur = s & 1;
#if TDM_OK
      asm volatile("s_wait_tensorcnt 0x0" ::: "memory");  // no-op on waves w/o TDM
#endif
      asm volatile("s_wait_asynccnt 0x0" ::: "memory");
      __syncthreads();
      if (s + 1 < nsteps)
        issue_tiles(A, W, K, Ncols, mBase, nBase, (s + 1) << 5,
                    Asf[cur ^ 1], Wsf[cur ^ 1], tid);
      gemm_step(Asf[cur], Wsf[cur], acc, wm, wn, lane);
      __syncthreads();
    }
  } else
#endif
  {
    // ---- guarded synchronous path (N tail blocks / fallback) ----
    for (int s = 0; s < nsteps; ++s) {
      stage_tiles_sync(A, W, K, Ncols, Mrows, mBase, nBase, s << 5,
                       Asf[0], Wsf[0], tid);
      __syncthreads();
      gemm_step(Asf[0], Wsf[0], acc, wm, wn, lane);
      __syncthreads();
    }
  }

  // ---- epilogue: C VGPR r holds (M = r + 8*(lane>=16), N = lane&15)
#pragma unroll
  for (int ti = 0; ti < 4; ++ti) {
#pragma unroll
    for (int tj = 0; tj < 2; ++tj) {
      int col = nBase + wn * 32 + tj * 16 + (lane & 15);
      if (col < Ncols) {
        float bv = bias[col];
#pragma unroll
        for (int r = 0; r < 8; ++r) {
          int row = mBase + wm * 64 + ti * 16 + ((lane >> 4) << 3) + r;
          if (row < Mrows) {
            float v = acc[ti][tj][r] + bv;
            if (relu) v = fmaxf(v, 0.f);
            C[(size_t)row * Ncols + col] = v;
          }
        }
      }
    }
  }
}

// =======================================================================
// Attention: one workgroup per (b, h).  N=64 queries, Mkv keys (64 or 196).
// Q/K/V/O are [tokens, D] with head h at columns h*64..h*64+63.
// =======================================================================
__global__ __launch_bounds__(256) void attn_kernel(
    const float* __restrict__ Q, const float* __restrict__ Kb,
    const float* __restrict__ Vb, const int* __restrict__ mask,
    float* __restrict__ O, int Mkv, int maskRowStride, int maskBatchStride) {
  __shared__ unsigned short qs[N_][HD_];        // 8 KB
  __shared__ unsigned short ks[MPMAX_][HD_];    // 28 KB
  __shared__ unsigned short vts[HD_][MPMAX_];   // 28 KB (V transposed)
  __shared__ float ss[N_][MPMAX_];              // 56 KB; reused in-place as bf16 P

  const int tid  = threadIdx.x;
  const int lane = tid & 31;
  const int w    = tid >> 5;
  const int b    = blockIdx.x / H_;
  const int h    = blockIdx.x % H_;
  const int Mp   = (Mkv + 31) & ~31;
  const size_t qbase = ((size_t)b * N_) * D_ + (size_t)h * HD_;
  const size_t kbase = ((size_t)b * Mkv) * D_ + (size_t)h * HD_;

  // ---- stage q, k, v(transposed) as bf16
  for (int idx = tid; idx < (N_ * HD_) / 2; idx += 256) {
    int n = idx >> 5, d = (idx & 31) * 2;
    float2 t = *(const float2*)&Q[qbase + (size_t)n * D_ + d];
    *(unsigned*)&qs[n][d] = pack2bf(t.x, t.y);
  }
  for (int idx = tid; idx < (Mp * HD_) / 2; idx += 256) {
    int m = idx >> 5, d = (idx & 31) * 2;
    unsigned pk = 0u;
    if (m < Mkv) {
      float2 t = *(const float2*)&Kb[kbase + (size_t)m * D_ + d];
      pk = pack2bf(t.x, t.y);
    }
    *(unsigned*)&ks[m][d] = pk;
  }
  for (int idx = tid; idx < Mp * HD_; idx += 256) {
    int m = idx >> 6, d = idx & 63;
    float v = (m < Mkv) ? Vb[kbase + (size_t)m * D_ + d] : 0.f;
    vts[d][m] = f2bf(v);
  }
  __syncthreads();

  // ---- scores S = (Q @ K^T) * scale, with mask
  const int ntiles = Mp >> 4;
  const int* mrow = mask + (size_t)b * maskBatchStride;
  for (int t = w; t < 4 * ntiles; t += 8) {
    int ti = t / ntiles, tj = t % ntiles;
    v8f acc = {};
#pragma unroll
    for (int kk = 0; kk < HD_; kk += 32) {
      v16bf a  = load_a_frag(&qs[0][0], ti * 16 + (lane & 15), HD_, kk, lane);
      v16bf bb = load_b_frag(&ks[0][0], tj * 16 + (lane & 15), HD_, kk, lane);
      acc = __builtin_amdgcn_wmma_f32_16x16x32_bf16(false, a, false, bb,
                                                    (short)0, acc, false, false);
    }
    int col = tj * 16 + (lane & 15);
#pragma unroll
    for (int r = 0; r < 8; ++r) {
      int row = ti * 16 + ((lane >> 4) << 3) + r;
      float s = acc[r] * SCALE_;
      int valid = (col < Mkv) ? mrow[row * maskRowStride + col] : 0;
      ss[row][col] = valid ? s : -1e30f;
    }
  }
  __syncthreads();

  // ---- softmax per row; write P as bf16 in place over ss (writes trail reads)
  unsigned short* ps = (unsigned short*)&ss[0][0];   // row stride 2*MPMAX_
  if (tid < N_) {
    int row = tid;
    float mx = -1e30f;
    for (int m = 0; m < Mkv; ++m) mx = fmaxf(mx, ss[row][m]);
    float sum = 0.f;
    for (int m = 0; m < Mkv; ++m) sum += expf(ss[row][m] - mx);
    float inv = 1.0f / sum;
    unsigned short* pr = ps + (size_t)row * (2 * MPMAX_);
    for (int m = 0; m < Mkv; ++m) {
      float e = expf(ss[row][m] - mx) * inv;
      pr[m] = f2bf(e);
    }
    for (int m = Mkv; m < Mp; ++m) pr[m] = 0;
  }
  __syncthreads();

  // ---- O = P @ V   (4 q-tiles x 4 d-tiles = 16 tiles over 8 waves)
  for (int t = w; t < 16; t += 8) {
    int ti = t >> 2, tj = t & 3;
    v8f acc = {};
    for (int kk = 0; kk < Mp; kk += 32) {
      v16bf a  = load_a_frag(ps, ti * 16 + (lane & 15), 2 * MPMAX_, kk, lane);
      v16bf bb = load_b_frag(&vts[0][0], tj * 16 + (lane & 15), MPMAX_, kk, lane);
      acc = __builtin_amdgcn_wmma_f32_16x16x32_bf16(false, a, false, bb,
                                                    (short)0, acc, false, false);
    }
    int col = tj * 16 + (lane & 15);
#pragma unroll
    for (int r = 0; r < 8; ++r) {
      int row = ti * 16 + ((lane >> 4) << 3) + r;
      O[qbase + (size_t)row * D_ + col] = acc[r];
    }
  }
}

// =======================================================================
// LayerNorm with fused residual: out = LN(Yin + R) * g + b ; D = 512
// =======================================================================
__global__ __launch_bounds__(256) void ln_kernel(
    const float* __restrict__ Yin, const float* __restrict__ R,
    const float* __restrict__ g, const float* __restrict__ bb,
    float* __restrict__ out) {
  __shared__ float red[256];
  const int row = blockIdx.x;
  const int t   = threadIdx.x;
  const size_t base = (size_t)row * D_;
  float a0 = Yin[base + t]       + R[base + t];
  float a1 = Yin[base + t + 256] + R[base + t + 256];

  red[t] = a0 + a1;
  __syncthreads();
  for (int s = 128; s > 0; s >>= 1) { if (t < s) red[t] += red[t + s]; __syncthreads(); }
  float mu = red[0] * (1.0f / D_);
  __syncthreads();

  float d0 = a0 - mu, d1 = a1 - mu;
  red[t] = d0 * d0 + d1 * d1;
  __syncthreads();
  for (int s = 128; s > 0; s >>= 1) { if (t < s) red[t] += red[t + s]; __syncthreads(); }
  float rs = rsqrtf(red[0] * (1.0f / D_) + 1e-5f);

  out[base + t]       = d0 * rs * g[t]       + bb[t];
  out[base + t + 256] = d1 * rs * g[t + 256] + bb[t + 256];
}

// =======================================================================
// Embedding + sinusoidal positions
// =======================================================================
__global__ __launch_bounds__(256) void embed_kernel(
    const int* __restrict__ seq, const float* __restrict__ emb,
    float* __restrict__ X) {
  int idx = blockIdx.x * 256 + threadIdx.x;
  if (idx >= TOK_ * D_) return;
  int d  = idx & (D_ - 1);
  int bn = idx >> 9;
  int n  = bn & (N_ - 1);
  int tok = seq[bn];
  float p = (float)(n + 1);
  int j = d >> 1;
  float ang = p * expf(-(float)(2 * j) * 9.210340371976184f / (float)D_);
  float pe  = (d & 1) ? cosf(ang) : sinf(ang);
  X[idx] = emb[(size_t)tok * D_ + d] + pe;
}

// =======================================================================
// host launcher
// =======================================================================
static inline void launch_gemm(const float* A, const float* W, const float* bias,
                               float* C, int Mr, int K, int Nc, int relu,
                               hipStream_t s) {
  dim3 grid((Nc + 127) / 128, (Mr + 127) / 128);
  gemm_bf16_kernel<<<grid, dim3(256), 0, s>>>(A, W, bias, C, Mr, K, Nc, relu);
}

extern "C" void kernel_launch(void* const* d_in, const int* in_sizes, int n_in,
                              void* d_out, int out_size, void* d_ws, size_t ws_size,
                              hipStream_t stream) {
  (void)in_sizes; (void)n_in; (void)out_size; (void)ws_size;
  const int*   seq      = (const int*)  d_in[0];
  const int*   seq_mask = (const int*)  d_in[1];
  const int*   att_mask = (const int*)  d_in[2];
  const float* enc      = (const float*)d_in[3];
  const float* word_emb = (const float*)d_in[4];
  const float* Wq  = (const float*)d_in[5];
  const float* Wk  = (const float*)d_in[6];
  const float* Wv  = (const float*)d_in[7];
  const float* Wo  = (const float*)d_in[8];
  const float* cWq = (const float*)d_in[9];
  const float* cWk = (const float*)d_in[10];
  const float* cWv = (const float*)d_in[11];
  const float* cWo = (const float*)d_in[12];
  const float* bq  = (const float*)d_in[13];
  const float* bk  = (const float*)d_in[14];
  const float* bv  = (const float*)d_in[15];
  const float* bo  = (const float*)d_in[16];
  const float* cbq = (const float*)d_in[17];
  const float* cbk = (const float*)d_in[18];
  const float* cbv = (const float*)d_in[19];
  const float* cbo = (const float*)d_in[20];
  const float* ln1_g = (const float*)d_in[21];
  const float* ln2_g = (const float*)d_in[22];
  const float* ln3_g = (const float*)d_in[23];
  const float* ln1_b = (const float*)d_in[24];
  const float* ln2_b = (const float*)d_in[25];
  const float* ln3_b = (const float*)d_in[26];
  const float* fc1_w = (const float*)d_in[27];
  const float* fc1_b = (const float*)d_in[28];
  const float* fc2_w = (const float*)d_in[29];
  const float* fc2_b = (const float*)d_in[30];
  const float* gen_w = (const float*)d_in[31];
  const float* gen_b = (const float*)d_in[32];

  // workspace layout (floats)
  float* X  = (float*)d_ws;                       // [TOK , D]
  float* Yb = X  + (size_t)TOK_  * D_;            // [TOK , D]
  float* Qb = Yb + (size_t)TOK_  * D_;            // [TOK , D]
  float* Kb = Qb + (size_t)TOK_  * D_;            // [ETOK, D]
  float* Vb = Kb + (size_t)ETOK_ * D_;            // [ETOK, D]
  float* Ab = Vb + (size_t)ETOK_ * D_;            // [TOK , D]
  float* Hh = Kb;                                 // [TOK, FF] aliases K/V (disjoint lifetime)

  embed_kernel<<<(TOK_ * D_ + 255) / 256, 256, 0, stream>>>(seq, word_emb, X);

  for (int i = 0; i < DEPTH_; ++i) {
    const size_t wo  = (size_t)i * D_ * D_;
    const size_t bo_ = (size_t)i * D_;
    // ---- self attention ----
    launch_gemm(X, Wq + wo, bq + bo_, Qb, TOK_, D_, D_, 0, stream);
    launch_gemm(X, Wk + wo, bk + bo_, Kb, TOK_, D_, D_, 0, stream);
    launch_gemm(X, Wv + wo, bv + bo_, Vb, TOK_, D_, D_, 0, stream);
    attn_kernel<<<B_ * H_, 256, 0, stream>>>(Qb, Kb, Vb, seq_mask, Ab,
                                             N_, N_, N_ * N_);
    launch_gemm(Ab, Wo + wo, bo + bo_, Yb, TOK_, D_, D_, 0, stream);
    ln_kernel<<<TOK_, 256, 0, stream>>>(Yb, X, ln1_g + bo_, ln1_b + bo_, X);
    // ---- cross attention ----
    launch_gemm(X,   cWq + wo, cbq + bo_, Qb, TOK_,  D_, D_, 0, stream);
    launch_gemm(enc, cWk + wo, cbk + bo_, Kb, ETOK_, D_, D_, 0, stream);
    launch_gemm(enc, cWv + wo, cbv + bo_, Vb, ETOK_, D_, D_, 0, stream);
    attn_kernel<<<B_ * H_, 256, 0, stream>>>(Qb, Kb, Vb, att_mask, Ab,
                                             M_, 0, M_);
    launch_gemm(Ab, cWo + wo, cbo + bo_, Yb, TOK_, D_, D_, 0, stream);
    ln_kernel<<<TOK_, 256, 0, stream>>>(Yb, X, ln2_g + bo_, ln2_b + bo_, X);
    // ---- FFN ----
    launch_gemm(X, fc1_w + (size_t)i * D_ * FF_, fc1_b + (size_t)i * FF_,
                Hh, TOK_, D_, FF_, 1, stream);
    launch_gemm(Hh, fc2_w + (size_t)i * FF_ * D_, fc2_b + bo_,
                Yb, TOK_, FF_, D_, 0, stream);
    ln_kernel<<<TOK_, 256, 0, stream>>>(Yb, X, ln3_g + bo_, ln3_b + bo_, X);
  }

  // generator: [TOK, D] @ [D, VOCAB] + gen_b -> d_out
  launch_gemm(X, gen_w, gen_b, (float*)d_out, TOK_, D_, VOC_, 0, stream);
}
